// LSTM_523986010344
// MI455X (gfx1250) — compile-verified, matrix operand order checked by reading, not run on hardware
//
#include <hip/hip_runtime.h>
#include <hip/hip_bf16.h>
#include <math.h>

// Problem constants (from reference): T=256, B=64, I=512, H=1024, O=256, G=4H=4096
#define TT 256
#define BB 64
#define II 512
#define HH 1024
#define OO 256
#define GG 4096

typedef __bf16 bf16;
typedef __attribute__((ext_vector_type(16))) __bf16 v16bf;
typedef __attribute__((ext_vector_type(8)))  float  v8f;

// ---------------------------------------------------------------------------
// Fragment loader for V_WMMA_F32_16X16X32_BF16.
// 16-bit A (16x32 MxK) layout (ISA 7.12.2): lanes 0-15 hold row M=lane,
// K = {k0..k0+7} and {k0+16..k0+23}; lanes 16-31 hold row M=lane-16,
// K = {k0+8..k0+15} and {k0+24..k0+31}. B (32x16 KxN) mirrors this with the
// lane indexing the N column of a row-major N x K weight matrix, so both A
// (M x K row-major) and B (N x K row-major) load with the same pattern:
// two contiguous 16-byte chunks per lane.
// ---------------------------------------------------------------------------
__device__ __forceinline__ v16bf load_frag(const bf16* base, int row0, int stride, int k0) {
  int lane = threadIdx.x & 31;
  int r    = lane & 15;
  int kh   = (lane >> 4) * 8;  // 0 for lanes 0-15, 8 for lanes 16-31
  const bf16* p = base + (size_t)(row0 + r) * stride + (k0 + kh);
  union { v16bf v; uint4 u[2]; } U;
  U.u[0] = *reinterpret_cast<const uint4*>(p);
  U.u[1] = *reinterpret_cast<const uint4*>(p + 16);
  return U.v;
}

// 3-term split-bf16 accumulation: (Ahi+Alo)(Bhi+Blo) ~= AhiBhi + AhiBlo + AloBhi
// (lo*lo term is ~2^-16 relative; dropped). Recovers ~f32 precision on the
// bf16 WMMA fast path.
__device__ __forceinline__ v8f wmma3(v16bf ah, v16bf al, v16bf bh, v16bf bl, v8f c) {
  c = __builtin_amdgcn_wmma_f32_16x16x32_bf16(false, ah, false, bh, (short)0, c, false, false);
  c = __builtin_amdgcn_wmma_f32_16x16x32_bf16(false, ah, false, bl, (short)0, c, false, false);
  c = __builtin_amdgcn_wmma_f32_16x16x32_bf16(false, al, false, bh, (short)0, c, false, false);
  return c;
}

__device__ __forceinline__ float sigf(float x) { return 1.0f / (1.0f + __expf(-x)); }

// ---------------------------------------------------------------------------
// f32 -> (bf16 hi, bf16 lo) split
// ---------------------------------------------------------------------------
__global__ void k_split(const float* __restrict__ in, bf16* __restrict__ hi,
                        bf16* __restrict__ lo, int n) {
  int i = blockIdx.x * blockDim.x + threadIdx.x;
  if (i < n) {
    float f = in[i];
    bf16  h = (bf16)f;
    hi[i] = h;
    lo[i] = (bf16)(f - (float)h);
  }
}

__global__ void k_init_state(bf16* __restrict__ h0h, bf16* __restrict__ h0l,
                             float* __restrict__ c) {
  int i = blockIdx.x * blockDim.x + threadIdx.x;
  if (i < BB * HH) {
    h0h[i] = (bf16)0.0f;
    h0l[i] = (bf16)0.0f;
    c[i]   = 0.0f;
  }
}

// ---------------------------------------------------------------------------
// xz = x @ W_ih^T + b_ih + b_hh   (M=T*B=16384, N=4H=4096, K=I=512)
// One wave computes a 16(M) x 64(N) tile (4 accumulators, A reused 4x).
// Grid: 65536 waves -> 8192 blocks x 256 threads.
// ---------------------------------------------------------------------------
__global__ __launch_bounds__(256) void k_xz_gemm(
    const bf16* __restrict__ xh, const bf16* __restrict__ xl,
    const bf16* __restrict__ wh, const bf16* __restrict__ wl,
    const float* __restrict__ b_ih, const float* __restrict__ b_hh,
    float* __restrict__ xz) {
  int wave  = (blockIdx.x * blockDim.x + threadIdx.x) >> 5;
  int nsup  = wave & 63;        // 64 super-tiles of 64 columns
  int mtile = wave >> 6;        // 1024 M tiles
  int mbase = mtile * 16;
  int nbase = nsup * 64;

  v8f acc[4] = {v8f{}, v8f{}, v8f{}, v8f{}};
  for (int k = 0; k < II; k += 32) {
    v16bf ah = load_frag(xh, mbase, II, k);
    v16bf al = load_frag(xl, mbase, II, k);
#pragma unroll
    for (int j = 0; j < 4; ++j) {
      int n0 = nbase + j * 16;
      v16bf bh = load_frag(wh, n0, II, k);
      v16bf bl = load_frag(wl, n0, II, k);
      acc[j] = wmma3(ah, al, bh, bl, acc[j]);
    }
  }
  // C/D layout: VGPR r -> M = mbase + r + (lane>=16 ? 8:0), N = nbase + (lane&15)
  int lane   = threadIdx.x & 31;
  int rowoff = (lane >> 4) * 8;
#pragma unroll
  for (int j = 0; j < 4; ++j) {
    int n = nbase + j * 16 + (lane & 15);
    float bias = b_ih[n] + b_hh[n];
#pragma unroll
    for (int r = 0; r < 8; ++r) {
      int row = mbase + rowoff + r;
      xz[(size_t)row * GG + n] = acc[j][r] + bias;
    }
  }
}

// ---------------------------------------------------------------------------
// One LSTM time step: z = xz_t + h @ W_hh^T ; gates ; c,h update.
// Wave owns (batch-tile m of 16 rows) x (16 hidden cols) x all 4 gates, so the
// c/h update is wave-local. h hi/lo is double-buffered across launches.
// Grid: 4 * 64 = 256 waves -> 32 blocks x 256 threads.
// ---------------------------------------------------------------------------
__global__ __launch_bounds__(256) void k_lstm_step(
    const float* __restrict__ xz_t,
    const bf16* __restrict__ hih, const bf16* __restrict__ hil,   // h in  (hi/lo)
    bf16* __restrict__ hoh, bf16* __restrict__ hol,               // h out (hi/lo)
    float* __restrict__ c,
    const bf16* __restrict__ whh_h, const bf16* __restrict__ whh_l,
    bf16* __restrict__ lh_h, bf16* __restrict__ lh_l,             // last-batch h trace
    int t) {
  int wave  = (blockIdx.x * blockDim.x + threadIdx.x) >> 5;  // 0..255
  int mtile = wave & 3;    // batch tile (B=64 -> 4 tiles)
  int ct    = wave >> 2;   // hidden column tile (H=1024 -> 64 tiles)
  int mbase = mtile * 16;
  int cbase = ct * 16;

  v8f acc[4] = {v8f{}, v8f{}, v8f{}, v8f{}};  // gates i,f,g,o
  for (int k = 0; k < HH; k += 32) {
    v16bf ah = load_frag(hih, mbase, HH, k);
    v16bf al = load_frag(hil, mbase, HH, k);
#pragma unroll
    for (int g = 0; g < 4; ++g) {
      int n0 = g * HH + cbase;   // W_hh row for gate g, hidden col tile
      v16bf bh = load_frag(whh_h, n0, HH, k);
      v16bf bl = load_frag(whh_l, n0, HH, k);
      acc[g] = wmma3(ah, al, bh, bl, acc[g]);
    }
  }

  int lane   = threadIdx.x & 31;
  int col    = cbase + (lane & 15);
  int rowoff = (lane >> 4) * 8;
#pragma unroll
  for (int r = 0; r < 8; ++r) {
    int row = mbase + rowoff + r;
    const float* xzr = xz_t + (size_t)row * GG;
    float zi = acc[0][r] + xzr[col];
    float zf = acc[1][r] + xzr[col + HH];
    float zg = acc[2][r] + xzr[col + 2 * HH];
    float zo = acc[3][r] + xzr[col + 3 * HH];
    float gi = sigf(zi);
    float gf = sigf(zf);
    float gg = tanhf(zg);
    float go = sigf(zo);
    size_t idx = (size_t)row * HH + col;
    float cn = gf * c[idx] + gi * gg;
    c[idx] = cn;
    float hval = go * tanhf(cn);
    bf16 hh = (bf16)hval;
    bf16 hl = (bf16)(hval - (float)hh);
    hoh[idx] = hh;
    hol[idx] = hl;
    if (row == BB - 1) {  // out[:, -1, :] trace for the final FC
      lh_h[(size_t)t * HH + col] = hh;
      lh_l[(size_t)t * HH + col] = hl;
    }
  }
}

// ---------------------------------------------------------------------------
// out = last @ fc_W^T + fc_b   (M=T=256, N=O=256, K=H=1024)
// 256 wave-tiles of 16x16 -> 32 blocks x 256 threads.
// ---------------------------------------------------------------------------
__global__ __launch_bounds__(256) void k_fc_gemm(
    const bf16* __restrict__ ah_, const bf16* __restrict__ al_,
    const bf16* __restrict__ wh, const bf16* __restrict__ wl,
    const float* __restrict__ bias, float* __restrict__ out) {
  int wave  = (blockIdx.x * blockDim.x + threadIdx.x) >> 5;  // 0..255
  int mtile = wave >> 4;   // T/16 = 16
  int ntile = wave & 15;   // O/16 = 16
  int mbase = mtile * 16;
  int nbase = ntile * 16;

  v8f acc = v8f{};
  for (int k = 0; k < HH; k += 32) {
    v16bf a_h = load_frag(ah_, mbase, HH, k);
    v16bf a_l = load_frag(al_, mbase, HH, k);
    v16bf b_h = load_frag(wh, nbase, HH, k);
    v16bf b_l = load_frag(wl, nbase, HH, k);
    acc = wmma3(a_h, a_l, b_h, b_l, acc);
  }
  int lane   = threadIdx.x & 31;
  int n      = nbase + (lane & 15);
  int rowoff = (lane >> 4) * 8;
  float b = bias[n];
#pragma unroll
  for (int r = 0; r < 8; ++r) {
    int row = mbase + rowoff + r;
    out[(size_t)row * OO + n] = acc[r] + b;
  }
}

// ---------------------------------------------------------------------------
// Host-side orchestration (graph-capture safe: stream-only, deterministic).
// Workspace (~330 MB): xz f32 + bf16 hi/lo splits of x, W_ih, W_hh, fc_W,
// double-buffered h hi/lo, c, last-h trace.
// ---------------------------------------------------------------------------
extern "C" void kernel_launch(void* const* d_in, const int* in_sizes, int n_in,
                              void* d_out, int out_size, void* d_ws, size_t ws_size,
                              hipStream_t stream) {
  (void)in_sizes; (void)n_in; (void)out_size; (void)ws_size;
  const float* x    = (const float*)d_in[0];  // (T,B,I)
  const float* W_ih = (const float*)d_in[1];  // (4H,I)
  const float* W_hh = (const float*)d_in[2];  // (4H,H)
  const float* b_ih = (const float*)d_in[3];  // (4H)
  const float* b_hh = (const float*)d_in[4];  // (4H)
  const float* fc_W = (const float*)d_in[5];  // (O,H)
  const float* fc_b = (const float*)d_in[6];  // (O)
  float* out = (float*)d_out;                 // (T,O)

  char* w = (char*)d_ws;
  auto alloc = [&](size_t bytes) -> char* {
    char* p = w;
    w += (bytes + 255) & ~(size_t)255;
    return p;
  };
  const size_t nx   = (size_t)TT * BB * II;  // 8.4M
  const size_t nwih = (size_t)GG * II;       // 2.1M
  const size_t nwhh = (size_t)GG * HH;       // 4.2M
  const size_t nfcw = (size_t)OO * HH;       // 0.26M
  const size_t nh   = (size_t)BB * HH;       // 64K
  const size_t nlh  = (size_t)TT * HH;       // 256K

  float* xz    = (float*)alloc((size_t)TT * BB * GG * 4);  // 256 MB
  bf16* x_hi   = (bf16*)alloc(nx * 2);
  bf16* x_lo   = (bf16*)alloc(nx * 2);
  bf16* wih_hi = (bf16*)alloc(nwih * 2);
  bf16* wih_lo = (bf16*)alloc(nwih * 2);
  bf16* whh_hi = (bf16*)alloc(nwhh * 2);
  bf16* whh_lo = (bf16*)alloc(nwhh * 2);
  bf16* fcw_hi = (bf16*)alloc(nfcw * 2);
  bf16* fcw_lo = (bf16*)alloc(nfcw * 2);
  bf16* hbuf_hi[2] = {(bf16*)alloc(nh * 2), (bf16*)alloc(nh * 2)};
  bf16* hbuf_lo[2] = {(bf16*)alloc(nh * 2), (bf16*)alloc(nh * 2)};
  float* cbuf  = (float*)alloc(nh * 4);
  bf16* lh_hi  = (bf16*)alloc(nlh * 2);
  bf16* lh_lo  = (bf16*)alloc(nlh * 2);

  // 1) split inputs to bf16 hi/lo
  k_split<<<(int)((nx   + 255) / 256), 256, 0, stream>>>(x,    x_hi,   x_lo,   (int)nx);
  k_split<<<(int)((nwih + 255) / 256), 256, 0, stream>>>(W_ih, wih_hi, wih_lo, (int)nwih);
  k_split<<<(int)((nwhh + 255) / 256), 256, 0, stream>>>(W_hh, whh_hi, whh_lo, (int)nwhh);
  k_split<<<(int)((nfcw + 255) / 256), 256, 0, stream>>>(fc_W, fcw_hi, fcw_lo, (int)nfcw);

  // 2) zero initial h (hi/lo) and c
  k_init_state<<<(int)((nh + 255) / 256), 256, 0, stream>>>(hbuf_hi[0], hbuf_lo[0], cbuf);

  // 3) big parallel GEMM: xz = x @ W_ih^T + biases
  k_xz_gemm<<<8192, 256, 0, stream>>>(x_hi, x_lo, wih_hi, wih_lo, b_ih, b_hh, xz);

  // 4) serial scan: 256 small step kernels (graph replay amortizes launches)
  for (int t = 0; t < TT; ++t) {
    int rb = t & 1, wb = rb ^ 1;
    k_lstm_step<<<32, 256, 0, stream>>>(
        xz + (size_t)t * BB * GG,
        hbuf_hi[rb], hbuf_lo[rb], hbuf_hi[wb], hbuf_lo[wb],
        cbuf, whh_hi, whh_lo, lh_hi, lh_lo, t);
  }

  // 5) final projection
  k_fc_gemm<<<32, 256, 0, stream>>>(lh_hi, lh_lo, fcw_hi, fcw_lo, fc_b, out);
}
